// TransLayer_13477607375001
// MI455X (gfx1250) — compile-verified, matrix operand order checked by reading, not run on hardware
//
#include <hip/hip_runtime.h>
#include <hip/hip_bf16.h>
#include <stdint.h>

// ---------------- types ----------------
typedef __attribute__((ext_vector_type(8)))  __bf16 v8bf;
typedef __attribute__((ext_vector_type(16))) __bf16 v16bf;
typedef __attribute__((ext_vector_type(8)))  float  v8f;

#define BATCH 4
#define SEQ   8192
#define DMODEL 512
#define HEADS 8
#define DH    64
#define LAND  256
#define BH    (BATCH*HEADS)      // 32
#define TOK   (BATCH*SEQ)        // 32768
#define QKVN  (3*DMODEL)         // 1536
#define KSZ   33

// ---------------- bf16 helpers (bit-level, no __bf16 arithmetic) ----------------
__device__ __forceinline__ float bf2f(unsigned short b) {
    return __uint_as_float(((unsigned)b) << 16);
}
__device__ __forceinline__ unsigned short f2bf(float f) {
    unsigned u = __float_as_uint(f);
    u += 0x7FFFu + ((u >> 16) & 1u);   // round-to-nearest-even
    return (unsigned short)(u >> 16);
}

// ---------------- WMMA fragment load ----------------
// 16-bit A/B 16x32 fragment layout (ISA 7.12.2): lane l holds row (l&15),
// K elements {8*(l>>4)+0..7} in halves [0..7] and {16+8*(l>>4)+0..7} in [8..15].
__device__ __forceinline__ v16bf load_frag(const unsigned short* p, int ld) {
    int lane = threadIdx.x & 31;
    int r    = lane & 15;
    int hi   = (lane >> 4) << 3;           // 0 or 8
    const __bf16* q = (const __bf16*)(p + (size_t)r * ld);
    v8bf lo = *(const v8bf*)(q + hi);
    v8bf h2 = *(const v8bf*)(q + 16 + hi);
    return __builtin_shufflevector(lo, h2, 0,1,2,3,4,5,6,7,8,9,10,11,12,13,14,15);
}

__device__ __forceinline__ void zero8(v8f& c) {
    for (int i = 0; i < 8; ++i) c[i] = 0.0f;
}

// core: D(32x32 per wave) = A(m0..m0+31, :) * B(n0..n0+31, :)^T, K-contiguous operands
__device__ __forceinline__ void gemm_core(const unsigned short* __restrict__ A, int lda,
                                          const unsigned short* __restrict__ B, int ldb,
                                          int K, int m0, int n0,
                                          v8f& c00, v8f& c01, v8f& c10, v8f& c11) {
    zero8(c00); zero8(c01); zero8(c10); zero8(c11);
    for (int kk = 0; kk < K; kk += 32) {
        // WGP-scope prefetch 4 K-tiles (128 elements) ahead (global_prefetch_b8, SCOPE 0:
        // pull into all cache levels so the consuming load hits the WGP$)
        __builtin_prefetch(A + (size_t)m0 * lda + kk + 128, 0, 3);
        __builtin_prefetch(B + (size_t)n0 * ldb + kk + 128, 0, 3);
        v16bf a0 = load_frag(A + (size_t)m0       * lda + kk, lda);
        v16bf a1 = load_frag(A + (size_t)(m0 + 16) * lda + kk, lda);
        v16bf b0 = load_frag(B + (size_t)n0       * ldb + kk, ldb);
        v16bf b1 = load_frag(B + (size_t)(n0 + 16) * ldb + kk, ldb);
        c00 = __builtin_amdgcn_wmma_f32_16x16x32_bf16(false, a0, false, b0, (short)0, c00, false, false);
        c01 = __builtin_amdgcn_wmma_f32_16x16x32_bf16(false, a0, false, b1, (short)0, c01, false, false);
        c10 = __builtin_amdgcn_wmma_f32_16x16x32_bf16(false, a1, false, b0, (short)0, c10, false, false);
        c11 = __builtin_amdgcn_wmma_f32_16x16x32_bf16(false, a1, false, b1, (short)0, c11, false, false);
    }
}

// C/D layout (ISA 7.12.2): lane: col = n0 + (lane&15); VGPR v: row = m0 + v + 8*(lane>>4)
__device__ __forceinline__ void store_acc_bf(unsigned short* __restrict__ D, int ldd,
                                             int m0, int n0, const v8f& c,
                                             float alpha, float beta) {
    int lane = threadIdx.x & 31;
    int col  = n0 + (lane & 15);
    int rb   = (lane >> 4) << 3;
    for (int v = 0; v < 8; ++v) {
        int row = m0 + rb + v;
        float val = alpha * c[v] + ((row == col) ? beta : 0.0f);
        D[(size_t)row * ldd + col] = f2bf(val);
    }
}

// ---------------- generic batched TN WMMA GEMM ----------------
// D[m,n] = alpha * sum_k A[m,k]*B[n,k] + beta*(m==n)
// block = 256 thr = 8 waves (4 along M x 2 along N): tile 128M x 64N
__global__ __launch_bounds__(256)
void k_gemm_tn(const unsigned short* __restrict__ A, int lda, long sA,
               const unsigned short* __restrict__ B, int ldb, long sB,
               unsigned short* __restrict__ D, int ldd, long sD,
               int K, float alpha, float beta) {
    int wave = threadIdx.x >> 5;
    int wm = wave & 3, wn = wave >> 2;
    int m0 = blockIdx.y * 128 + wm * 32;
    int n0 = blockIdx.x * 64  + wn * 32;
    int z  = blockIdx.z;
    const unsigned short* Az = A + (size_t)z * sA;
    const unsigned short* Bz = B + (size_t)z * sB;
    unsigned short* Dz = D + (size_t)z * sD;
    v8f c00, c01, c10, c11;
    gemm_core(Az, lda, Bz, ldb, K, m0, n0, c00, c01, c10, c11);
    store_acc_bf(Dz, ldd, m0,      n0,      c00, alpha, beta);
    store_acc_bf(Dz, ldd, m0,      n0 + 16, c01, alpha, beta);
    store_acc_bf(Dz, ldd, m0 + 16, n0,      c10, alpha, beta);
    store_acc_bf(Dz, ldd, m0 + 16, n0 + 16, c11, alpha, beta);
}

// ---------------- QKV GEMM: xn(32768x512) @ w_qkv^T(512x1536) with head split ----------------
__global__ __launch_bounds__(256)
void k_gemm_qkv(const unsigned short* __restrict__ A,   // xn bf16, ld 512
                const unsigned short* __restrict__ B,   // w_qkv bf16 (1536x512)
                unsigned short* __restrict__ Q,
                unsigned short* __restrict__ Kd,
                unsigned short* __restrict__ V) {
    int wave = threadIdx.x >> 5;
    int wm = wave & 3, wn = wave >> 2;
    int m0 = blockIdx.y * 128 + wm * 32;
    int n0 = blockIdx.x * 64  + wn * 32;
    v8f c[4];
    gemm_core(A, DMODEL, B, DMODEL, DMODEL, m0, n0, c[0], c[1], c[2], c[3]);
    int lane = threadIdx.x & 31;
    int rb = (lane >> 4) << 3;
    for (int t = 0; t < 4; ++t) {
        int mi = (t >> 1) * 16, ni = (t & 1) * 16;
        int col = n0 + ni + (lane & 15);
        int part = col >> 9;            // 0=q 1=k 2=v
        int hh   = (col >> 6) & 7;
        int cc   = col & 63;
        unsigned short* dst = (part == 0) ? Q : ((part == 1) ? Kd : V);
        float scale = (part == 0) ? 0.125f : 1.0f;  // dh^-0.5
        for (int v = 0; v < 8; ++v) {
            int row = m0 + mi + rb + v;             // global token
            int bb = row >> 13, nn = row & 8191;
            size_t o = ((((size_t)bb * 8 + hh) * SEQ + nn) << 6) + cc;
            dst[o] = f2bf(c[t][v] * scale);
        }
    }
}

// ---------------- final GEMM: y = x + out_bnd @ w_out^T + b_out (fp32 out) ----------------
__global__ __launch_bounds__(256)
void k_gemm_final(const unsigned short* __restrict__ A,  // out_bnd bf16 (32768x512)
                  const unsigned short* __restrict__ B,  // w_out bf16 (512x512)
                  const float* __restrict__ x,
                  const float* __restrict__ bias,
                  float* __restrict__ out) {
    int wave = threadIdx.x >> 5;
    int wm = wave & 3, wn = wave >> 2;
    int m0 = blockIdx.y * 128 + wm * 32;
    int n0 = blockIdx.x * 64  + wn * 32;
    v8f c[4];
    gemm_core(A, DMODEL, B, DMODEL, DMODEL, m0, n0, c[0], c[1], c[2], c[3]);
    int lane = threadIdx.x & 31;
    int rb = (lane >> 4) << 3;
    for (int t = 0; t < 4; ++t) {
        int mi = (t >> 1) * 16, ni = (t & 1) * 16;
        int col = n0 + ni + (lane & 15);
        for (int v = 0; v < 8; ++v) {
            int row = m0 + mi + rb + v;
            size_t o = (size_t)row * DMODEL + col;
            out[o] = c[t][v] + bias[col] + x[o];
        }
    }
}

// ---------------- fused-attention building block ----------------
// scores tile: sS[32x256] = A(32xDH) @ B(256xDH)^T, both K-contiguous bf16.
// 8 waves: wm = wave&1 (16-row group), wn = wave>>1 (64-col group), 4 accs/wave.
__device__ __forceinline__ void scores_32x256(const unsigned short* __restrict__ A,
                                              const unsigned short* __restrict__ B,
                                              float* __restrict__ sS) {
    int wave = threadIdx.x >> 5;
    int wm = wave & 1, wn = wave >> 1;
    int m0 = wm * 16, n0 = wn * 64;
    v8f acc[4];
    for (int ni = 0; ni < 4; ++ni) zero8(acc[ni]);
    for (int kk = 0; kk < DH; kk += 32) {
        v16bf a = load_frag(A + (size_t)m0 * DH + kk, DH);
#pragma unroll
        for (int ni = 0; ni < 4; ++ni) {
            v16bf b = load_frag(B + (size_t)(n0 + ni * 16) * DH + kk, DH);
            acc[ni] = __builtin_amdgcn_wmma_f32_16x16x32_bf16(false, a, false, b, (short)0, acc[ni], false, false);
        }
    }
    int lane = threadIdx.x & 31;
    int rb = (lane >> 4) << 3, cl = lane & 15;
#pragma unroll
    for (int ni = 0; ni < 4; ++ni)
        for (int v = 0; v < 8; ++v)
            sS[(m0 + rb + v) * 256 + n0 + ni * 16 + cl] = acc[ni][v];
}

// ---------------- fused attn1: out = softmax(q @ k_l^T) @ Z2 -> (b,n,h*dh) ----------------
// block: 32 query rows; LDS: fp32 scores 32KB + bf16 P 16KB. Zero global score traffic.
__global__ __launch_bounds__(256)
void k_attn1_fused(const unsigned short* __restrict__ q,    // (BH,SEQ,DH)
                   const unsigned short* __restrict__ kl,   // (BH,LAND,DH)
                   const unsigned short* __restrict__ z2T,  // (BH,DH,LAND)
                   unsigned short* __restrict__ outbnd) {   // (B,SEQ,DMODEL)
    __shared__ float sS[32 * 256];
    __shared__ unsigned short sP[32 * 256];
    __shared__ float sT[256];
    __shared__ float sM[32], sL[32];
    int bh = blockIdx.y;
    int tok0 = blockIdx.x * 32;
    const unsigned short* Aq = q + ((size_t)bh * SEQ + tok0) * DH;
    const unsigned short* Bk = kl + (size_t)bh * LAND * DH;
    const unsigned short* Bz = z2T + (size_t)bh * DH * LAND;
    int tid = threadIdx.x;
    scores_32x256(Aq, Bk, sS);
    __syncthreads();
    // parallel row softmax: 8 threads per row, 32 cols each
    int row = tid >> 3, part = tid & 7;
    float* rS = sS + row * 256 + part * 32;
    float mx = -3.4e38f;
    for (int j = 0; j < 32; ++j) mx = fmaxf(mx, rS[j]);
    sT[tid] = mx;
    __syncthreads();
    if (part == 0) {
        float m = sT[row * 8];
        for (int p = 1; p < 8; ++p) m = fmaxf(m, sT[row * 8 + p]);
        sM[row] = m;
    }
    __syncthreads();
    float m = sM[row], ps = 0.0f;
    for (int j = 0; j < 32; ++j) { float e = __expf(rS[j] - m); rS[j] = e; ps += e; }
    sT[tid] = ps;
    __syncthreads();
    if (part == 0) {
        float s = 0.0f;
        for (int p = 0; p < 8; ++p) s += sT[row * 8 + p];
        sL[row] = 1.0f / s;
    }
    __syncthreads();
    float inv = sL[row];
    unsigned short* rP = sP + row * 256 + part * 32;
    for (int j = 0; j < 32; ++j) rP[j] = f2bf(rS[j] * inv);
    __syncthreads();
    // GEMM2: (32x64) = P(32x256) @ Z2(256x64); A frags from LDS, B = z2T rows (K-contig)
    int wave = tid >> 5;
    int m0 = (wave & 1) * 16, n0 = (wave >> 1) * 16;
    v8f o; zero8(o);
    for (int kk = 0; kk < LAND; kk += 32) {
        v16bf a = load_frag(sP + (size_t)m0 * LAND + kk, LAND);
        v16bf b = load_frag(Bz + (size_t)n0 * LAND + kk, LAND);
        o = __builtin_amdgcn_wmma_f32_16x16x32_bf16(false, a, false, b, (short)0, o, false, false);
    }
    int lane = tid & 31, rb = (lane >> 4) << 3, cl = lane & 15;
    int bb = bh >> 3, hh = bh & 7;
    for (int v = 0; v < 8; ++v) {
        int tok = tok0 + m0 + rb + v;
        outbnd[((size_t)bb * SEQ + tok) * DMODEL + hh * DH + n0 + cl] = f2bf(o[v]);
    }
}

// ---------------- fused attn3: W3V = softmax(q_l @ k^T) @ v ----------------
// block: 32 landmark rows, two-pass online softmax over 8192 keys in 256-key chunks.
__global__ __launch_bounds__(256)
void k_attn3_fused(const unsigned short* __restrict__ ql,   // (BH,LAND,DH)
                   const unsigned short* __restrict__ kb,   // (BH,SEQ,DH)
                   const unsigned short* __restrict__ vT,   // (BH,DH,SEQ)
                   unsigned short* __restrict__ w3v) {      // (BH,LAND,DH)
    __shared__ float sS[32 * 256];
    __shared__ unsigned short sP[32 * 256];
    __shared__ float sT[256];
    __shared__ float sM[32], sL[32];
    int bh = blockIdx.y;
    int r0 = blockIdx.x * 32;
    const unsigned short* Aq = ql + ((size_t)bh * LAND + r0) * DH;
    const unsigned short* Kz = kb + (size_t)bh * SEQ * DH;
    const unsigned short* Vz = vT + (size_t)bh * DH * SEQ;
    int tid = threadIdx.x;
    int row = tid >> 3, part = tid & 7;
    if (tid < 32) { sM[tid] = -3.4e38f; sL[tid] = 0.0f; }
    __syncthreads();
    // ---- pass 1: running (max, sumexp) per row ----
    for (int c0 = 0; c0 < SEQ; c0 += 256) {
        scores_32x256(Aq, Kz + (size_t)c0 * DH, sS);
        __syncthreads();
        const float* rS = sS + row * 256 + part * 32;
        float mx = -3.4e38f;
        for (int j = 0; j < 32; ++j) mx = fmaxf(mx, rS[j]);
        sT[tid] = mx;
        __syncthreads();
        if (part == 0) {
            float cm = sT[row * 8];
            for (int p = 1; p < 8; ++p) cm = fmaxf(cm, sT[row * 8 + p]);
            float om = sM[row];
            float nm = fmaxf(om, cm);
            sL[row] *= __expf(om - nm);    // first chunk: 0 * exp(-inf) = 0
            sM[row] = nm;
        }
        __syncthreads();
        float m = sM[row], ps = 0.0f;
        for (int j = 0; j < 32; ++j) ps += __expf(rS[j] - m);
        sT[tid] = ps;
        __syncthreads();
        if (part == 0) {
            float s = 0.0f;
            for (int p = 0; p < 8; ++p) s += sT[row * 8 + p];
            sL[row] += s;
        }
        __syncthreads();
    }
    if (tid < 32) sL[tid] = 1.0f / sL[tid];
    __syncthreads();
    // ---- pass 2: O += P_chunk @ V_chunk (persistent WMMA accumulator) ----
    int wave = tid >> 5;
    int m0 = (wave & 1) * 16, n0 = (wave >> 1) * 16;   // n0: dh column group
    v8f o; zero8(o);
    for (int c0 = 0; c0 < SEQ; c0 += 256) {
        scores_32x256(Aq, Kz + (size_t)c0 * DH, sS);
        __syncthreads();
        {
            float m = sM[row], inv = sL[row];
            const float* rS = sS + row * 256 + part * 32;
            unsigned short* rP = sP + row * 256 + part * 32;
            for (int j = 0; j < 32; ++j) rP[j] = f2bf(__expf(rS[j] - m) * inv);
        }
        __syncthreads();
        for (int kk = 0; kk < 256; kk += 32) {
            v16bf a = load_frag(sP + (size_t)m0 * 256 + kk, 256);
            v16bf b = load_frag(Vz + (size_t)n0 * SEQ + c0 + kk, SEQ);
            o = __builtin_amdgcn_wmma_f32_16x16x32_bf16(false, a, false, b, (short)0, o, false, false);
        }
        __syncthreads();
    }
    int lane = tid & 31, rb = (lane >> 4) << 3, cl = lane & 15;
    for (int v = 0; v < 8; ++v) {
        int rr = r0 + m0 + rb + v;
        w3v[((size_t)bh * LAND + rr) * DH + n0 + cl] = f2bf(o[v]);
    }
}

// ---------------- LayerNorm -> bf16 ----------------
__global__ __launch_bounds__(256)
void k_ln(const float* __restrict__ x, const float* __restrict__ w,
          const float* __restrict__ b, unsigned short* __restrict__ xn) {
    __shared__ float s1[256], s2[256];
    int row = blockIdx.x, t = threadIdx.x;
    const float* xr = x + (size_t)row * DMODEL;
    float a = xr[t], c = xr[t + 256];
    s1[t] = a + c; s2[t] = a * a + c * c;
    __syncthreads();
    for (int s = 128; s > 0; s >>= 1) {
        if (t < s) { s1[t] += s1[t + s]; s2[t] += s2[t + s]; }
        __syncthreads();
    }
    float mu = s1[0] * (1.0f / DMODEL);
    float var = s2[0] * (1.0f / DMODEL) - mu * mu;
    float rstd = rsqrtf(var + 1e-5f);
    size_t base = (size_t)row * DMODEL;
    xn[base + t]       = f2bf((a - mu) * rstd * w[t] + b[t]);
    xn[base + t + 256] = f2bf((c - mu) * rstd * w[t + 256] + b[t + 256]);
}

// ---------------- misc elementwise ----------------
__global__ void k_cvt(const float* __restrict__ in, unsigned short* __restrict__ out, int n) {
    int i = blockIdx.x * 256 + threadIdx.x;
    if (i < n) out[i] = f2bf(in[i]);
}

// landmark mean over groups of 32 tokens
__global__ void k_landmark(const unsigned short* __restrict__ src, unsigned short* __restrict__ dst) {
    int bh = blockIdx.x >> 8, mm = blockIdx.x & 255;
    int c = threadIdx.x;                      // 64 threads
    const unsigned short* p = src + (((size_t)bh * SEQ + (size_t)mm * 32) << 6) + c;
    float s = 0.0f;
    for (int t = 0; t < 32; ++t) s += bf2f(p[(size_t)t * DH]);
    dst[(((size_t)bh * LAND + mm) << 6) + c] = f2bf(s * (1.0f / 32.0f));
}

// in-place bf16 row softmax, contiguous rows of length L (used for attn2 only)
__global__ __launch_bounds__(256)
void k_softmax(unsigned short* __restrict__ S, int L) {
    __shared__ float red[256];
    size_t base = (size_t)blockIdx.x * L;
    int t = threadIdx.x;
    float mx = -3.4e38f;
    for (int i = t; i < L; i += 256) mx = fmaxf(mx, bf2f(S[base + i]));
    red[t] = mx; __syncthreads();
    for (int s = 128; s > 0; s >>= 1) { if (t < s) red[t] = fmaxf(red[t], red[t + s]); __syncthreads(); }
    mx = red[0]; __syncthreads();
    float sum = 0.0f;
    for (int i = t; i < L; i += 256) {
        float e = __expf(bf2f(S[base + i]) - mx);
        sum += e;
        S[base + i] = f2bf(e);
    }
    red[t] = sum; __syncthreads();
    for (int s = 128; s > 0; s >>= 1) { if (t < s) red[t] += red[t + s]; __syncthreads(); }
    float inv = 1.0f / red[0];
    for (int i = t; i < L; i += 256) S[base + i] = f2bf(bf2f(S[base + i]) * inv);
}

// batched bf16 transpose: out[c*rows+r] = in[r*cols+c], rows/cols multiples of 32
__global__ __launch_bounds__(256)
void k_transpose(const unsigned short* __restrict__ in, unsigned short* __restrict__ out,
                 int rows, int cols) {
    __shared__ unsigned short tile[32][33];
    size_t zoff = (size_t)blockIdx.z * rows * cols;
    int c0 = blockIdx.x * 32, r0 = blockIdx.y * 32;
    int tx = threadIdx.x & 31, ty = threadIdx.x >> 5;
    for (int i = ty; i < 32; i += 8)
        tile[i][tx] = in[zoff + (size_t)(r0 + i) * cols + c0 + tx];
    __syncthreads();
    for (int i = ty; i < 32; i += 8)
        out[zoff + (size_t)(c0 + i) * rows + r0 + tx] = tile[tx][i];
}

// ---------------- pinv helpers ----------------
__global__ void k_init_scale(int* scl) { scl[0] = 0; scl[1] = 0; }

__global__ __launch_bounds__(256)
void k_absmax(const unsigned short* __restrict__ X, int* __restrict__ scl) {
    __shared__ float r1[256], r2[256];
    int t = threadIdx.x;
    const unsigned short* Xb = X + ((size_t)blockIdx.x << 16);
    float rs = 0.0f, cs = 0.0f;
    for (int j = 0; j < 256; ++j) rs += fabsf(bf2f(Xb[t * 256 + j]));
    for (int i = 0; i < 256; ++i) cs += fabsf(bf2f(Xb[i * 256 + t]));
    r1[t] = rs; r2[t] = cs; __syncthreads();
    for (int s = 128; s > 0; s >>= 1) {
        if (t < s) { r1[t] = fmaxf(r1[t], r1[t + s]); r2[t] = fmaxf(r2[t], r2[t + s]); }
        __syncthreads();
    }
    if (t == 0) {
        atomicMax(&scl[0], __float_as_int(r1[0]));
        atomicMax(&scl[1], __float_as_int(r2[0]));
    }
}

// Z0 = X^T / (max_rowsum * max_colsum)
__global__ void k_make_z0(const unsigned short* __restrict__ X, unsigned short* __restrict__ Z,
                          const int* __restrict__ scl) {
    float inv = 1.0f / (__int_as_float(scl[0]) * __int_as_float(scl[1]));
    size_t idx = (size_t)blockIdx.x * 256 + threadIdx.x;   // over 32*256*256
    size_t z = idx >> 16;
    int e = (int)(idx & 65535);
    int i = e >> 8, j = e & 255;
    Z[(z << 16) + (size_t)j * 256 + i] = f2bf(bf2f(X[(z << 16) + (size_t)i * 256 + j]) * inv);
}

// T1 = 7I - XZ (elementwise)
__global__ void k_ew7(const unsigned short* __restrict__ XZ, unsigned short* __restrict__ T1) {
    size_t idx = (size_t)blockIdx.x * 256 + threadIdx.x;
    int e = (int)(idx & 65535);
    int i = e >> 8, j = e & 255;
    T1[idx] = f2bf(((i == j) ? 7.0f : 0.0f) - bf2f(XZ[idx]));
}

// ---------------- depthwise conv residual add into out_bnd ----------------
__global__ __launch_bounds__(256)
void k_conv_add(const unsigned short* __restrict__ v, const float* __restrict__ cw,
                unsigned short* __restrict__ out_bnd) {
    size_t idx = (size_t)blockIdx.x * 256 + threadIdx.x;   // over BH*SEQ*DH
    int c = (int)(idx & 63);
    size_t r = idx >> 6;
    int n = (int)(r & 8191);
    int bh = (int)(r >> 13);
    int h = bh & 7;
    const unsigned short* vb = v + (((size_t)bh * SEQ) << 6) + c;
    float s = 0.0f;
    for (int t = 0; t < KSZ; ++t) {
        int nn = n + t - (KSZ / 2);
        if (nn >= 0 && nn < SEQ) s += cw[h * KSZ + t] * bf2f(vb[(size_t)nn * DH]);
    }
    size_t o = ((size_t)((bh >> 3) * SEQ + n)) * DMODEL + h * DH + c;
    out_bnd[o] = f2bf(bf2f(out_bnd[o]) + s);
}

// ---------------- host side ----------------
extern "C" void kernel_launch(void* const* d_in, const int* in_sizes, int n_in,
                              void* d_out, int out_size, void* d_ws, size_t ws_size,
                              hipStream_t stream) {
    const float* x      = (const float*)d_in[0];
    const float* ln_w   = (const float*)d_in[1];
    const float* ln_b   = (const float*)d_in[2];
    const float* w_qkv  = (const float*)d_in[3];
    const float* w_out  = (const float*)d_in[4];
    const float* b_out  = (const float*)d_in[5];
    const float* conv_w = (const float*)d_in[6];
    float* out = (float*)d_out;

    // ---- workspace carve (bf16 buffers stored as u16) ----
    char* ws = (char*)d_ws;
    size_t off = 0;
    auto alloc = [&](size_t bytes) -> void* {
        void* p = ws + off;
        off += (bytes + 255) & ~(size_t)255;
        return p;
    };
    typedef unsigned short u16;
    u16* xn     = (u16*)alloc((size_t)TOK * DMODEL * 2);
    u16* wqkv_b = (u16*)alloc((size_t)QKVN * DMODEL * 2);
    u16* wout_b = (u16*)alloc((size_t)DMODEL * DMODEL * 2);
    u16* qb     = (u16*)alloc((size_t)BH * SEQ * DH * 2);
    u16* kb     = (u16*)alloc((size_t)BH * SEQ * DH * 2);
    u16* vb     = (u16*)alloc((size_t)BH * SEQ * DH * 2);
    u16* ql     = (u16*)alloc((size_t)BH * LAND * DH * 2);
    u16* kl     = (u16*)alloc((size_t)BH * LAND * DH * 2);
    u16* X2     = (u16*)alloc((size_t)BH * LAND * LAND * 2);   // attn2
    u16* XZ     = (u16*)alloc((size_t)BH * LAND * LAND * 2);
    u16* T1     = (u16*)alloc((size_t)BH * LAND * LAND * 2);
    u16* T2     = (u16*)alloc((size_t)BH * LAND * LAND * 2);
    u16* TT     = (u16*)alloc((size_t)BH * LAND * LAND * 2);   // transpose staging
    u16* Zc     = (u16*)alloc((size_t)BH * LAND * LAND * 2);
    u16* Zo     = (u16*)alloc((size_t)BH * LAND * LAND * 2);
    int* scl    = (int*)alloc(256);
    u16* vT     = (u16*)alloc((size_t)BH * DH * SEQ * 2);
    u16* w3v    = (u16*)alloc((size_t)BH * LAND * DH * 2);
    u16* w3vT   = (u16*)alloc((size_t)BH * DH * LAND * 2);
    u16* z2     = (u16*)alloc((size_t)BH * LAND * DH * 2);
    u16* z2T    = (u16*)alloc((size_t)BH * DH * LAND * 2);
    u16* outbnd = (u16*)alloc((size_t)TOK * DMODEL * 2);
    (void)ws_size; (void)in_sizes; (void)n_in; (void)out_size;

    const long sMM = (long)LAND * LAND;          // 65536
    const long sQL = (long)LAND * DH;            // 16384

    // ---- weights -> bf16, layernorm ----
    k_cvt<<<(QKVN * DMODEL + 255) / 256, 256, 0, stream>>>(w_qkv, wqkv_b, QKVN * DMODEL);
    k_cvt<<<(DMODEL * DMODEL + 255) / 256, 256, 0, stream>>>(w_out, wout_b, DMODEL * DMODEL);
    k_ln<<<TOK, 256, 0, stream>>>(x, ln_w, ln_b, xn);

    // ---- QKV projection (WMMA) ----
    k_gemm_qkv<<<dim3(QKVN / 64, TOK / 128, 1), 256, 0, stream>>>(xn, wqkv_b, qb, kb, vb);

    // ---- landmarks ----
    k_landmark<<<BH * LAND, 64, 0, stream>>>(qb, ql);
    k_landmark<<<BH * LAND, 64, 0, stream>>>(kb, kl);

    // ---- attn2 = softmax(q_l @ k_l^T) ----
    k_gemm_tn<<<dim3(LAND / 64, LAND / 128, BH), 256, 0, stream>>>(
        ql, DH, sQL, kl, DH, sQL, X2, LAND, sMM, DH, 1.0f, 0.0f);
    k_softmax<<<BH * LAND, 256, 0, stream>>>(X2, LAND);

    // ---- pinv init: Z = X^T / (max rowsum * max colsum) ----
    k_init_scale<<<1, 1, 0, stream>>>(scl);
    k_absmax<<<BH, 256, 0, stream>>>(X2, scl);
    k_make_z0<<<(int)(BH * sMM / 256), 256, 0, stream>>>(X2, Zc, scl);

    // ---- 6 Newton iterations: Z = 0.25*Z*(13I - XZ*(15I - XZ*(7I - XZ))) ----
    dim3 gMM(LAND / 64, LAND / 128, BH);
    dim3 gTR(LAND / 32, LAND / 32, BH);
    int nEW = (int)(BH * sMM / 256);
    for (int it = 0; it < 6; ++it) {
        k_transpose<<<gTR, 256, 0, stream>>>(Zc, TT, LAND, LAND);
        k_gemm_tn<<<gMM, 256, 0, stream>>>(X2, LAND, sMM, TT, LAND, sMM, XZ, LAND, sMM,
                                           LAND, 1.0f, 0.0f);
        k_ew7<<<nEW, 256, 0, stream>>>(XZ, T1);
        k_transpose<<<gTR, 256, 0, stream>>>(T1, TT, LAND, LAND);
        k_gemm_tn<<<gMM, 256, 0, stream>>>(XZ, LAND, sMM, TT, LAND, sMM, T2, LAND, sMM,
                                           LAND, -1.0f, 15.0f);
        k_transpose<<<gTR, 256, 0, stream>>>(T2, TT, LAND, LAND);
        k_gemm_tn<<<gMM, 256, 0, stream>>>(XZ, LAND, sMM, TT, LAND, sMM, T1, LAND, sMM,
                                           LAND, -1.0f, 13.0f);   // T1 = T3
        k_transpose<<<gTR, 256, 0, stream>>>(T1, TT, LAND, LAND);
        k_gemm_tn<<<gMM, 256, 0, stream>>>(Zc, LAND, sMM, TT, LAND, sMM, Zo, LAND, sMM,
                                           LAND, 0.25f, 0.0f);
        u16* tmp = Zc; Zc = Zo; Zo = tmp;
    }

    // ---- fused attn3: W3V = softmax(q_l @ k^T) @ v  (needs v^T) ----
    k_transpose<<<dim3(DH / 32, SEQ / 32, BH), 256, 0, stream>>>(vb, vT, SEQ, DH);
    k_attn3_fused<<<dim3(LAND / 32, BH), 256, 0, stream>>>(ql, kb, vT, w3v);

    // ---- Z2 = pinv(attn2) @ W3V  (associativity: avoids n x m x m GEMM) ----
    k_transpose<<<dim3(DH / 32, LAND / 32, BH), 256, 0, stream>>>(w3v, w3vT, LAND, DH);
    k_gemm_tn<<<dim3(DH / 64, LAND / 128, BH), 256, 0, stream>>>(
        Zc, LAND, sMM, w3vT, LAND, sQL, z2, DH, sQL, LAND, 1.0f, 0.0f);

    // ---- fused attn1: out = softmax(q @ k_l^T) @ Z2 -> (b,n,h*dh) ----
    k_transpose<<<dim3(DH / 32, LAND / 32, BH), 256, 0, stream>>>(z2, z2T, LAND, DH);
    k_attn1_fused<<<dim3(SEQ / 32, BH), 256, 0, stream>>>(qb, kl, z2T, outbnd);

    // ---- residual depthwise conv on v, add into out ----
    k_conv_add<<<(int)((size_t)BH * SEQ * DH / 256), 256, 0, stream>>>(vb, conv_w, outbnd);

    // ---- final projection: y = x + out @ w_out^T + b_out ----
    k_gemm_final<<<dim3(DMODEL / 64, TOK / 128, 1), 256, 0, stream>>>(
        outbnd, wout_b, x, b_out, out);
}
